// PointSpatialConv_34205119545795
// MI455X (gfx1250) — matrix operand order repaired and checked by reading, not compiled, not run on hardware
//
#include <hip/hip_runtime.h>

#define BB 2
#define TT 8
#define NN 4096
#define MM 1024
#define KK 32
#define C1 64
#define C2 128
#define R2 0.81f
#define BNEPS 1e-5f

typedef __attribute__((ext_vector_type(16))) __bf16 v16bf;
typedef __attribute__((ext_vector_type(8)))  float  v8f;

__device__ __forceinline__ unsigned short f2bf(float f) {
  unsigned u = __float_as_uint(f);
  u += 0x7FFFu + ((u >> 16) & 1u);          // round-to-nearest-even
  return (unsigned short)(u >> 16);
}

// ---------------- init: W2 -> bf16, zero BN stat accumulators ----------------
__global__ void init_kernel(const float* __restrict__ W2,
                            unsigned short* __restrict__ w2bf,
                            float* __restrict__ stats1,
                            float* __restrict__ stats2) {
  int tid = threadIdx.x;
  for (int i = tid; i < C2 * C1; i += 256) w2bf[i] = f2bf(W2[i]);
  for (int i = tid; i < TT * C1 * 2; i += 256) stats1[i] = 0.f;
  for (int i = tid; i < TT * C2 * 2; i += 256) stats2[i] = 0.f;
}

// ---------------- FPS: one 256-thread block per frame ----------------
__global__ void __launch_bounds__(256)
fps_kernel(const float* __restrict__ xyzs, float* __restrict__ out_ref) {
  __shared__ float sd[256];
  __shared__ int   si[256];
  __shared__ float sl[3];
  int f = blockIdx.x;
  int tid = threadIdx.x;
  const float* base = xyzs + (size_t)f * NN * 3;

  float dist[16];
#pragma unroll
  for (int i = 0; i < 16; ++i) dist[i] = 1e10f;

  int last = 0;
  for (int it = 0; it < MM; ++it) {
    if (tid == 0) {
      const float* p = base + 3 * (size_t)last;
      float* o = out_ref + ((size_t)f * MM + it) * 3;
      o[0] = p[0]; o[1] = p[1]; o[2] = p[2];
      sl[0] = p[0]; sl[1] = p[1]; sl[2] = p[2];
    }
    __syncthreads();
    if (it == MM - 1) break;
    float lx = sl[0], ly = sl[1], lz = sl[2];
    float bd = -1.f; int bi = 0;
#pragma unroll
    for (int i = 0; i < 16; ++i) {
      int n = i * 256 + tid;
      const float* p = base + 3 * (size_t)n;
      float ddx = p[0] - lx, ddy = p[1] - ly, ddz = p[2] - lz;
      float d = ddx * ddx + ddy * ddy + ddz * ddz;
      dist[i] = fminf(dist[i], d);
      if (dist[i] > bd) { bd = dist[i]; bi = n; }   // ascending n -> first max
    }
    sd[tid] = bd; si[tid] = bi;
    __syncthreads();
    for (int s = 128; s > 0; s >>= 1) {
      if (tid < s) {
        float od = sd[tid + s]; int oi = si[tid + s];
        if (od > sd[tid] || (od == sd[tid] && oi < si[tid])) { sd[tid] = od; si[tid] = oi; }
      }
      __syncthreads();
    }
    last = si[0];
  }
}

// ---------------- ball query: one wave32 per reference point ----------------
__global__ void __launch_bounds__(256)
ballq_kernel(const float* __restrict__ xyzs, const float* __restrict__ ref,
             int* __restrict__ idxb) {
  int tid = threadIdx.x;
  int w = blockIdx.x * 8 + (tid >> 5);
  int lane = tid & 31;
  int f = w / MM;
  const float* base = xyzs + (size_t)f * NN * 3;
  float rx = ref[(size_t)w * 3 + 0];
  float ry = ref[(size_t)w * 3 + 1];
  float rz = ref[(size_t)w * 3 + 2];
  int* out = idxb + (size_t)w * KK;
  int cnt = 0, firstn = 0;
  bool have_first = false;
  for (int b0 = 0; b0 < NN; b0 += 32) {
    int n = b0 + lane;
    const float* p = base + 3 * (size_t)n;
    float dx = p[0] - rx, dy = p[1] - ry, dz = p[2] - rz;
    bool pred = (dx * dx + dy * dy + dz * dz) < R2;
    unsigned mask = __builtin_amdgcn_ballot_w32(pred);
    if (!have_first && mask) { firstn = b0 + __ffs(mask) - 1; have_first = true; }
    if (pred) {
      int slot = cnt + __popc(mask & ((1u << lane) - 1u));
      if (slot < KK) out[slot] = n;
    }
    cnt += __popc(mask);
    if (cnt >= KK) break;              // cnt uniform across wave
  }
  if (cnt < KK) {
    for (int s = cnt + lane; s < KK; s += 32) out[s] = firstn;
  }
}

// ---------------- MLP1 (C=4 -> 64) + BN1 stat accumulation ----------------
__global__ void __launch_bounds__(256)
mlp1_kernel(const float* __restrict__ xyzs, const float* __restrict__ ref,
            const int* __restrict__ idxb, const float* __restrict__ W1,
            unsigned short* __restrict__ h1, float* __restrict__ stats1) {
  __shared__ float dxs[KK], dys[KK], dzs[KK];
  __shared__ float ssum[256], ssq[256];
  int w = blockIdx.x;
  int tid = threadIdx.x;
  int f = w / MM;
  int t = f % TT;
  const float* base = xyzs + (size_t)f * NN * 3;
  if (tid < KK) {
    float rx = ref[(size_t)w * 3 + 0];
    float ry = ref[(size_t)w * 3 + 1];
    float rz = ref[(size_t)w * 3 + 2];
    int n = idxb[(size_t)w * KK + tid];
    const float* p = base + 3 * (size_t)n;
    dxs[tid] = p[0] - rx; dys[tid] = p[1] - ry; dzs[tid] = p[2] - rz;
  }
  __syncthreads();
  int c1 = tid & 63, kg = tid >> 6;
  float w0 = W1[c1 * 4 + 0], w1v = W1[c1 * 4 + 1], w2v = W1[c1 * 4 + 2];
  unsigned short* hb = h1 + (size_t)w * (KK * C1);
  float ls = 0.f, lq = 0.f;
#pragma unroll
  for (int kk = 0; kk < 8; ++kk) {
    int k = kg * 8 + kk;
    float h = dxs[k] * w0 + dys[k] * w1v + dzs[k] * w2v;
    hb[k * C1 + c1] = f2bf(h);
    ls += h; lq += h * h;
  }
  ssum[tid] = ls; ssq[tid] = lq;
  __syncthreads();
  if (tid < 64) {
    float S = ssum[tid] + ssum[tid + 64] + ssum[tid + 128] + ssum[tid + 192];
    float Q = ssq[tid] + ssq[tid + 64] + ssq[tid + 128] + ssq[tid + 192];
    atomicAdd(&stats1[(t * C1 + tid) * 2 + 0], S);
    atomicAdd(&stats1[(t * C1 + tid) * 2 + 1], Q);
  }
}

// ---------------- BN finalize: stats -> per-channel scale/bias ----------------
__global__ void bn_fin_kernel(const float* __restrict__ stats,
                              const float* __restrict__ g, const float* __restrict__ b,
                              float* __restrict__ scale, float* __restrict__ bias,
                              int C) {
  int i = threadIdx.x;
  if (i < TT * C) {
    float n = (float)(BB * MM * KK);
    float mean = stats[i * 2] / n;
    float var  = stats[i * 2 + 1] / n - mean * mean;
    int c = i % C;
    float s = g[c] * rsqrtf(var + BNEPS);
    scale[i] = s;
    bias[i]  = b[c] - mean * s;
  }
}

// ---------------- BN1 + ReLU applied in place to h1 (bf16) ----------------
// One uint4 (8 bf16 elems) per thread; h1 layout (w, k, c1): 8 elems share k,
// c1 = (quad % 8) * 8 .. +7.
__global__ void __launch_bounds__(256)
bn1_apply_kernel(uint4* __restrict__ h1q, const float* __restrict__ scale1,
                 const float* __restrict__ bias1) {
  unsigned q = blockIdx.x * 256 + threadIdx.x;    // 16384*256 quads total
  int w = q >> 8;                                  // 256 quads per reference point
  int t = (w >> 10) & (TT - 1);
  const float* sc = scale1 + t * C1;
  const float* bi = bias1 + t * C1;
  int c0 = (q & 7) * 8;
  uint4 v = h1q[q];
  unsigned r[4] = {v.x, v.y, v.z, v.w};
#pragma unroll
  for (int j = 0; j < 4; ++j) {
    int c = c0 + 2 * j;
    float lo = __uint_as_float(r[j] << 16);
    float hi = __uint_as_float(r[j] & 0xFFFF0000u);
    lo = fmaxf(fmaf(lo, sc[c],     bi[c]),     0.f);
    hi = fmaxf(fmaf(hi, sc[c + 1], bi[c + 1]), 0.f);
    r[j] = (unsigned)f2bf(lo) | ((unsigned)f2bf(hi) << 16);
  }
  h1q[q] = make_uint4(r[0], r[1], r[2], r[3]);
}

// ---------------- WMMA fragment builders (contiguous k order, same in A and B) ----------------
// A 16x32 bf16, lane row = lane&15: pairs {half*8+0..7} (K = 16*half .. +15)
// -> 8 contiguous dwords => two global_load_b128.
__device__ __forceinline__ v16bf load_A(const unsigned* __restrict__ row,
                                        int half, int kstep) {
  union { v16bf v; unsigned u[8]; } r;
  const unsigned* q = row + kstep * 16;
#pragma unroll
  for (int j = 0; j < 8; ++j) r.u[j] = q[half * 8 + j];
  return r.v;
}
// B 32x16 bf16, lane col = lane&15: lanes 0-15 hold K=0-15, lanes 16-31 K=16-31
// -> 8 contiguous dwords => two global_load_b128.
__device__ __forceinline__ v16bf load_B(const unsigned* __restrict__ w2u,
                                        int c2, int half, int kstep) {
  union { v16bf v; unsigned u[8]; } r;
  const unsigned* row = w2u + c2 * 32 + kstep * 16 + half * 8;
#pragma unroll
  for (int j = 0; j < 8; ++j) r.u[j] = row[j];
  return r.v;
}

#define WMMA_BF16(a, b, c) \
  __builtin_amdgcn_wmma_f32_16x16x32_bf16(false, (a), false, (b), (short)0, (c), false, false)

// ---------------- MLP2 pass 1: WMMA GEMM, accumulate BN2 stats ----------------
__global__ void __launch_bounds__(256)
mlp2_stats_kernel(const unsigned short* __restrict__ h1,
                  const unsigned short* __restrict__ w2bf,
                  float* __restrict__ stats2) {
  __shared__ float s2sum[C2], s2sq[C2];
  int tid = threadIdx.x;
  if (tid < C2) { s2sum[tid] = 0.f; s2sq[tid] = 0.f; }
  __syncthreads();

  int w = blockIdx.x * 8 + (tid >> 5);
  int lane = tid & 31, half = lane >> 4, c2l = lane & 15;
  int t = (w / MM) % TT;
  const unsigned* h1u = (const unsigned*)(h1 + (size_t)w * (KK * C1));
  const unsigned* w2u = (const unsigned*)w2bf;

  int klo = lane & 15;
  const unsigned* r0 = h1u + klo * 32;          // row tile 0 (k = 0..15)
  const unsigned* r1 = h1u + (16 + klo) * 32;   // row tile 1 (k = 16..31)
  v16bf a00 = load_A(r0, half, 0), a01 = load_A(r0, half, 1);
  v16bf a10 = load_A(r1, half, 0), a11 = load_A(r1, half, 1);

#pragma unroll
  for (int nt = 0; nt < 8; ++nt) {
    int c2 = nt * 16 + c2l;
    v16bf b0 = load_B(w2u, c2, half, 0);
    v16bf b1 = load_B(w2u, c2, half, 1);
    v8f c = {}; v8f d = {};
    c = WMMA_BF16(a00, b0, c);
    d = WMMA_BF16(a10, b0, d);      // independent chain hides c's RAW hazard
    c = WMMA_BF16(a01, b1, c);
    d = WMMA_BF16(a11, b1, d);
    float ps = 0.f, pq = 0.f;
#pragma unroll
    for (int r = 0; r < 8; ++r) {
      float vc = c[r]; ps += vc; pq += vc * vc;
      float vd = d[r]; ps += vd; pq += vd * vd;
    }
    atomicAdd(&s2sum[c2], ps);
    atomicAdd(&s2sq[c2], pq);
  }
  __syncthreads();
  if (tid < C2) {
    atomicAdd(&stats2[(t * C2 + tid) * 2 + 0], s2sum[tid]);
    atomicAdd(&stats2[(t * C2 + tid) * 2 + 1], s2sq[tid]);
  }
}

// ---------------- MLP2 pass 2: WMMA GEMM + BN2 + ReLU + max over K ----------------
__global__ void __launch_bounds__(256)
mlp2_final_kernel(const unsigned short* __restrict__ h1,
                  const unsigned short* __restrict__ w2bf,
                  const float* __restrict__ scale2, const float* __restrict__ bias2,
                  float* __restrict__ feat) {
  int tid = threadIdx.x;
  int w = blockIdx.x * 8 + (tid >> 5);
  int lane = tid & 31, half = lane >> 4, c2l = lane & 15;
  int bt = w / MM, m = w % MM, t = bt % TT;
  const unsigned* h1u = (const unsigned*)(h1 + (size_t)w * (KK * C1));
  const unsigned* w2u = (const unsigned*)w2bf;
  const float* s2 = scale2 + t * C2;
  const float* b2 = bias2 + t * C2;

  int klo = lane & 15;
  const unsigned* r0 = h1u + klo * 32;
  const unsigned* r1 = h1u + (16 + klo) * 32;
  v16bf a00 = load_A(r0, half, 0), a01 = load_A(r0, half, 1);
  v16bf a10 = load_A(r1, half, 0), a11 = load_A(r1, half, 1);

#pragma unroll
  for (int nt = 0; nt < 8; ++nt) {
    int c2 = nt * 16 + c2l;
    v16bf b0 = load_B(w2u, c2, half, 0);
    v16bf b1 = load_B(w2u, c2, half, 1);
    v8f c = {}; v8f d = {};
    c = WMMA_BF16(a00, b0, c);
    d = WMMA_BF16(a10, b0, d);
    c = WMMA_BF16(a01, b1, c);
    d = WMMA_BF16(a11, b1, d);
    float s = s2[c2], bb = b2[c2];
    float cm = 0.f;                               // relu'd values are >= 0
#pragma unroll
    for (int r = 0; r < 8; ++r) {
      cm = fmaxf(cm, fmaxf(fmaf(c[r], s, bb), 0.f));
      cm = fmaxf(cm, fmaxf(fmaf(d[r], s, bb), 0.f));
    }
    float v = fmaxf(cm, __shfl_xor(cm, 16, 32));  // combine half-waves (rows +8)
    if (lane < 16) feat[((size_t)bt * C2 + c2) * MM + m] = v;
  }
}

// ---------------- launch ----------------
extern "C" void kernel_launch(void* const* d_in, const int* in_sizes, int n_in,
                              void* d_out, int out_size, void* d_ws, size_t ws_size,
                              hipStream_t stream) {
  (void)in_sizes; (void)n_in; (void)out_size; (void)ws_size;
  const float* xyzs = (const float*)d_in[0];
  const float* W1   = (const float*)d_in[1];
  const float* g1   = (const float*)d_in[2];
  const float* b1   = (const float*)d_in[3];
  const float* W2   = (const float*)d_in[4];
  const float* g2   = (const float*)d_in[5];
  const float* b2   = (const float*)d_in[6];

  float* out_ref  = (float*)d_out;
  float* out_feat = out_ref + (size_t)BB * TT * MM * 3;

  char* p = (char*)d_ws;
  int* idxb            = (int*)p;             p += (size_t)BB * TT * MM * KK * sizeof(int);
  unsigned short* h1   = (unsigned short*)p;  p += (size_t)BB * TT * MM * KK * C1 * sizeof(unsigned short);
  unsigned short* w2bf = (unsigned short*)p;  p += (size_t)C2 * C1 * sizeof(unsigned short);
  float* stats1        = (float*)p;           p += (size_t)TT * C1 * 2 * sizeof(float);
  float* stats2        = (float*)p;           p += (size_t)TT * C2 * 2 * sizeof(float);
  float* scale1        = (float*)p;           p += (size_t)TT * C1 * sizeof(float);
  float* bias1         = (float*)p;           p += (size_t)TT * C1 * sizeof(float);
  float* scale2        = (float*)p;           p += (size_t)TT * C2 * sizeof(float);
  float* bias2         = (float*)p;           p += (size_t)TT * C2 * sizeof(float);

  const int waves = BB * TT * MM;  // 16384 reference points, one wave each

  init_kernel<<<1, 256, 0, stream>>>(W2, w2bf, stats1, stats2);
  fps_kernel<<<BB * TT, 256, 0, stream>>>(xyzs, out_ref);
  ballq_kernel<<<waves / 8, 256, 0, stream>>>(xyzs, out_ref, idxb);
  mlp1_kernel<<<waves, 256, 0, stream>>>(xyzs, out_ref, idxb, W1, h1, stats1);
  bn_fin_kernel<<<1, 512, 0, stream>>>(stats1, g1, b1, scale1, bias1, C1);
  bn1_apply_kernel<<<waves, 256, 0, stream>>>((uint4*)h1, scale1, bias1);
  mlp2_stats_kernel<<<waves / 8, 256, 0, stream>>>(h1, w2bf, stats2);
  bn_fin_kernel<<<1, 1024, 0, stream>>>(stats2, g2, b2, scale2, bias2, C2);
  mlp2_final_kernel<<<waves / 8, 256, 0, stream>>>(h1, w2bf, scale2, bias2, out_feat);
}
